// TokenAdaptivePruner_84550726189538
// MI455X (gfx1250) — matrix operand order repaired and checked by reading, not compiled, not run on hardware
//
#include <hip/hip_runtime.h>

typedef __bf16 bf16;
typedef __attribute__((ext_vector_type(16))) __bf16 v16bf;
typedef __attribute__((ext_vector_type(8)))  __bf16 v8bf;
typedef __attribute__((ext_vector_type(8)))  float  v8f;

#define BB 32
#define LL 11
#define NN 577
#define CC 768
#define HH 12
#define HD 64
#define MTOK (BB * NN)   // 18464 = 577 * 32
#define NPAD 592         // 37 * 16   (K-rows / Q-rows padded)
#define NPADV 608        // 19 * 32   (P/V K-dim padded)
#define NKT 37

#define DEV static __device__ __forceinline__

DEV bf16 f2bf(float f) {
  unsigned u = __builtin_bit_cast(unsigned, f);
  unsigned r = u + 0x7fffu + ((u >> 16) & 1u);   // round-to-nearest-even
  unsigned short h = (unsigned short)(r >> 16);
  return __builtin_bit_cast(bf16, h);
}
DEV float bf2f(bf16 b) {
  unsigned short s = __builtin_bit_cast(unsigned short, b);
  unsigned u = ((unsigned)s) << 16;
  return __builtin_bit_cast(float, u);
}

DEV v8f wmma_bf16(v16bf a, v16bf b, v8f c) {
  // D = A(16x32) * B(32x16) + C, f32 accumulate
  return __builtin_amdgcn_wmma_f32_16x16x32_bf16(
      /*neg_a=*/false, a, /*neg_b=*/false, b,
      /*c_mod=*/(short)0, c, /*reuse_a=*/false, /*reuse_b=*/false);
}

// A-matrix 16x32 bf16 fragment: lane r (0-15) holds row r, K {0..7,16..23};
// lane r+16 holds row r, K {8..15,24..31}. Two 16B chunks per lane.
DEV v16bf load_a_frag(const bf16* p, int stride, int row0, int k0, int lane) {
  int r = lane & 15, hv = lane >> 4;
  const bf16* base = p + (size_t)(row0 + r) * stride + k0 + hv * 8;
  v8bf lo = *(const v8bf*)base;
  v8bf hc = *(const v8bf*)(base + 16);
  v16bf o;
#pragma unroll
  for (int i = 0; i < 8; ++i) { o[i] = lo[i]; o[i + 8] = hc[i]; }
  return o;
}

// B-matrix 32x16 bf16 fragment: lane n (0-15) holds column n, K {0..7,16..23};
// lane n+16 holds column n, K {8..15,24..31}. Column n sourced from row (col0+n)
// of a row-major [N,K] matrix (i.e. computes A * W^T).
DEV v16bf load_b_frag(const bf16* p, int stride, int col0, int k0, int lane) {
  int n = lane & 15, hv = lane >> 4;
  const bf16* base = p + (size_t)(col0 + n) * stride + k0 + hv * 8;
  v8bf lo = *(const v8bf*)base;
  v8bf hc = *(const v8bf*)(base + 16);
  v16bf o;
#pragma unroll
  for (int i = 0; i < 8; ++i) { o[i] = lo[i]; o[i + 8] = hc[i]; }
  return o;
}

DEV float gelu_exact(float x) {
  return 0.5f * x * (1.0f + erff(x * 0.7071067811865475f));
}

// ---------------------------------------------------------------- weights
__global__ void f32_to_bf16_kernel(const float* __restrict__ in,
                                   bf16* __restrict__ out, int n) {
  int i = blockIdx.x * 256 + threadIdx.x;
  if (i < n) out[i] = f2bf(in[i]);
}

// ---------------------------------------------------------------- prune
__global__ void __launch_bounds__(256) prune_kernel(
    const float* __restrict__ scores, float* __restrict__ maskf) {
  __shared__ float ps[576];
  __shared__ float val[1024];
  __shared__ float cs[1024];
  __shared__ int   ind[1024];
  __shared__ float red[256];
  const int b = blockIdx.x, tid = threadIdx.x;
  for (int i = tid; i < 576; i += 256) ps[i] = 0.f;
  __syncthreads();
  // softmax over patches per block-row, mean over LL rows
  for (int l = 0; l < LL; ++l) {
    const float* row = scores + ((size_t)b * LL + l) * NN + 1;
    float pm = -3.4e38f;
    for (int i = tid; i < 576; i += 256) { float v = row[i]; val[i] = v; pm = fmaxf(pm, v); }
    red[tid] = pm; __syncthreads();
    for (int st = 128; st > 0; st >>= 1) { if (tid < st) red[tid] = fmaxf(red[tid], red[tid + st]); __syncthreads(); }
    float mx = red[0]; __syncthreads();
    float pe = 0.f;
    for (int i = tid; i < 576; i += 256) pe += __expf(val[i] - mx);
    red[tid] = pe; __syncthreads();
    for (int st = 128; st > 0; st >>= 1) { if (tid < st) red[tid] += red[tid + st]; __syncthreads(); }
    float inv = 1.f / red[0]; __syncthreads();
    for (int i = tid; i < 576; i += 256) ps[i] += __expf(val[i] - mx) * inv * (1.f / (float)LL);
    __syncthreads();
  }
  // bitonic sort ascending over 1024 (pad with +inf)
  for (int i = tid; i < 1024; i += 256) { val[i] = (i < 576) ? ps[i] : 3.4e38f; ind[i] = i; }
  __syncthreads();
  for (int k = 2; k <= 1024; k <<= 1) {
    for (int j = k >> 1; j > 0; j >>= 1) {
      for (int i = tid; i < 1024; i += 256) {
        int ixj = i ^ j;
        if (ixj > i) {
          bool up = ((i & k) == 0);
          float a = val[i], c = val[ixj];
          bool sw = up ? (a > c) : (a < c);
          if (sw) {
            val[i] = c; val[ixj] = a;
            int t = ind[i]; ind[i] = ind[ixj]; ind[ixj] = t;
          }
        }
      }
      __syncthreads();
    }
  }
  // inclusive prefix sum (Hillis-Steele)
  for (int off = 1; off < 1024; off <<= 1) {
    for (int i = tid; i < 1024; i += 256) cs[i] = val[i] + ((i >= off) ? val[i - off] : 0.f);
    __syncthreads();
    for (int i = tid; i < 1024; i += 256) val[i] = cs[i];
    __syncthreads();
  }
  for (int i = tid; i < 576; i += 256)
    maskf[(size_t)b * NN + 1 + ind[i]] = (val[i] > 0.5f) ? 1.f : 0.f;
  if (tid == 0) maskf[(size_t)b * NN] = 1.f;   // CLS always kept
}

// ---------------------------------------------------------------- layernorm
__global__ void __launch_bounds__(256) ln_kernel(
    const float* __restrict__ x, const float* __restrict__ g,
    const float* __restrict__ bta, bf16* __restrict__ outB,
    float* __restrict__ outF) {
  __shared__ float red[256];
  const int row = blockIdx.x, tid = threadIdx.x;
  const float* xr = x + (size_t)row * CC;
  float s = 0.f, s2 = 0.f;
  for (int c = tid; c < CC; c += 256) { float v = xr[c]; s += v; s2 += v * v; }
  red[tid] = s; __syncthreads();
  for (int st = 128; st > 0; st >>= 1) { if (tid < st) red[tid] += red[tid + st]; __syncthreads(); }
  float mean = red[0] / (float)CC; __syncthreads();
  red[tid] = s2; __syncthreads();
  for (int st = 128; st > 0; st >>= 1) { if (tid < st) red[tid] += red[tid + st]; __syncthreads(); }
  float var = red[0] / (float)CC - mean * mean;
  float inv = rsqrtf(var + 1e-6f);
  for (int c = tid; c < CC; c += 256) {
    float v = (xr[c] - mean) * inv * g[c] + bta[c];
    if (outB) outB[(size_t)row * CC + c] = f2bf(v);
    if (outF) outF[(size_t)row * CC + c] = v;
  }
}

// ---------------------------------------------------------------- GEMM (WMMA)
// out[M, Nout] = A[M,K](bf16) * W[Nout,K]^T(bf16) + bias (+gelu) (+res)
// block = 4 waves (128 thr); wave -> 32x64 tile (2x4 WMMA); block -> 32x256.
// A-fragments are amortized over 4 B-tiles: 8 WMMA per 12 b128 loads.
__global__ void __launch_bounds__(128) gemm_bf16_kernel(
    const bf16* __restrict__ A, const bf16* __restrict__ W,
    const float* __restrict__ bias, const float* __restrict__ res,
    float* __restrict__ outF, bf16* __restrict__ outB,
    int Nout, int K, int doGelu) {
  const int wave = threadIdx.x >> 5, lane = threadIdx.x & 31;
  const int r = lane & 15, hv = lane >> 4;
  const int m0 = blockIdx.y * 32;
  const int n0 = blockIdx.x * 256 + wave * 64;
  const v8f z = {0.f, 0.f, 0.f, 0.f, 0.f, 0.f, 0.f, 0.f};
  v8f acc[2][4];
#pragma unroll
  for (int mt = 0; mt < 2; ++mt)
#pragma unroll
    for (int nt = 0; nt < 4; ++nt) acc[mt][nt] = z;

  for (int k0 = 0; k0 < K; k0 += 32) {
    // speculative prefetch 2 k-tiles ahead (global_prefetch_b8, no counter)
    if (k0 + 64 < K) {
      __builtin_prefetch(A + (size_t)(m0 + r + hv * 16) * K + k0 + 64, 0, 1);
      __builtin_prefetch(W + (size_t)(n0 + r + hv * 16) * K + k0 + 64, 0, 1);
    }
    v16bf a0 = load_a_frag(A, K, m0,      k0, lane);
    v16bf a1 = load_a_frag(A, K, m0 + 16, k0, lane);
    v16bf b[4];
#pragma unroll
    for (int nt = 0; nt < 4; ++nt) b[nt] = load_b_frag(W, K, n0 + nt * 16, k0, lane);
#pragma unroll
    for (int nt = 0; nt < 4; ++nt) {
      acc[0][nt] = wmma_bf16(a0, b[nt], acc[0][nt]);
      acc[1][nt] = wmma_bf16(a1, b[nt], acc[1][nt]);
    }
  }
#pragma unroll
  for (int mt = 0; mt < 2; ++mt)
#pragma unroll
    for (int nt = 0; nt < 4; ++nt) {
      int col = n0 + nt * 16 + r;
      float bv = bias ? bias[col] : 0.f;
#pragma unroll
      for (int i = 0; i < 8; ++i) {
        int rowi = m0 + mt * 16 + hv * 8 + i;
        float v = acc[mt][nt][i] + bv;
        if (doGelu) v = gelu_exact(v);
        size_t idx = (size_t)rowi * Nout + col;
        if (res)  v += res[idx];
        if (outF) outF[idx] = v;
        if (outB) outB[idx] = f2bf(v);
      }
    }
}

// ---------------------------------------------------------------- attn prep
// qkv_bf: [MTOK, 3*CC]. Emit per (b,h):
//   Qb [NPAD,64] (scaled by HD^-0.5), Kb [NPAD,64], VTb [64,NPADV] (V^T)
__global__ void __launch_bounds__(256) prep_attn_kernel(
    const bf16* __restrict__ qkv, bf16* __restrict__ Qb,
    bf16* __restrict__ Kb, bf16* __restrict__ VTb) {
  const int bh = blockIdx.x, b = bh / HH, h = bh % HH;
  bf16* Q  = Qb  + (size_t)bh * NPAD * HD;
  bf16* Kp = Kb  + (size_t)bh * NPAD * HD;
  bf16* VT = VTb + (size_t)bh * HD * NPADV;
  const bf16 z = f2bf(0.f);
  for (int t = threadIdx.x; t < NPAD * HD; t += 256) {
    int n = t / HD, d = t % HD;
    bf16 qv = z, kv = z;
    if (n < NN) {
      size_t base = ((size_t)(b * NN + n)) * (3 * CC) + h * HD + d;
      qv = f2bf(bf2f(qkv[base]) * 0.125f);   // HD^-0.5 = 1/8
      kv = qkv[base + CC];
    }
    Q[t] = qv; Kp[t] = kv;
  }
  for (int t = threadIdx.x; t < HD * NPADV; t += 256) {
    int d = t / NPADV, n = t % NPADV;
    bf16 vv = z;
    if (n < NN) vv = qkv[((size_t)(b * NN + n)) * (3 * CC) + 2 * CC + h * HD + d];
    VT[t] = vv;
  }
}

// ---------------------------------------------------------------- attention
// block: (q-tile, h, b); 128 threads = 4 waves.
__global__ void __launch_bounds__(128) attn_kernel(
    const bf16* __restrict__ Qb, const bf16* __restrict__ Kb,
    const bf16* __restrict__ VTb, const float* __restrict__ maskf,
    bf16* __restrict__ y_bf) {
  __shared__ __attribute__((aligned(16))) float S[16][NPAD];
  __shared__ __attribute__((aligned(16))) bf16  P[16][NPADV];
  __shared__ float mk[NPADV];
  __shared__ float red[16][8];
  const int qt = blockIdx.x, h = blockIdx.y, b = blockIdx.z;
  const int tid = threadIdx.x, wave = tid >> 5, lane = tid & 31;
  const int r = lane & 15, hv = lane >> 4;
  const bf16* Q  = Qb  + (size_t)(b * HH + h) * NPAD * HD;
  const bf16* Kp = Kb  + (size_t)(b * HH + h) * NPAD * HD;
  const bf16* VT = VTb + (size_t)(b * HH + h) * HD * NPADV;

  for (int c = tid; c < NPADV; c += 128) mk[c] = (c < NN) ? maskf[(size_t)b * NN + c] : 0.f;

  // phase 1: S = (Q * scale) K^T  -> LDS
  const v8f z = {0.f, 0.f, 0.f, 0.f, 0.f, 0.f, 0.f, 0.f};
  v16bf aq0 = load_a_frag(Q, HD, qt * 16, 0, lane);
  v16bf aq1 = load_a_frag(Q, HD, qt * 16, 32, lane);
  for (int kt = wave; kt < NKT; kt += 4) {
    v16bf bk0 = load_b_frag(Kp, HD, kt * 16, 0, lane);
    v16bf bk1 = load_b_frag(Kp, HD, kt * 16, 32, lane);
    v8f c = z;
    c = wmma_bf16(aq0, bk0, c);
    c = wmma_bf16(aq1, bk1, c);
#pragma unroll
    for (int i = 0; i < 8; ++i) S[hv * 8 + i][kt * 16 + r] = c[i];
  }
  __syncthreads();

  // phase 2: masked softmax per row, write bf16 P
  const int row = tid >> 3, sub = tid & 7;
  const int q = qt * 16 + row;
  const float mq = (q < NN) ? maskf[(size_t)b * NN + q] : 0.f;
  float pmax = -3.4e38f;
  for (int c = sub; c < NN; c += 8)
    if (mk[c] > 0.f) pmax = fmaxf(pmax, S[row][c]);
  red[row][sub] = pmax; __syncthreads();
  if (sub == 0) {
    float m = red[row][0];
#pragma unroll
    for (int i = 1; i < 8; ++i) m = fmaxf(m, red[row][i]);
    red[row][0] = m;
  }
  __syncthreads();
  float rmax = red[row][0];
  __syncthreads();
  float psum = 0.f;
  for (int c = sub; c < NN; c += 8)
    if (mk[c] > 0.f) psum += __expf(S[row][c] - rmax);
  red[row][sub] = psum; __syncthreads();
  if (sub == 0) {
    float ssum = 0.f;
#pragma unroll
    for (int i = 0; i < 8; ++i) ssum += red[row][i];
    red[row][0] = ssum;
  }
  __syncthreads();
  float rsum = red[row][0];
  float inv = (mq > 0.f && rsum > 0.f) ? (1.f / rsum) : 0.f;
  for (int c = sub; c < NPADV; c += 8) {
    float p = 0.f;
    if (inv != 0.f && c < NN && mk[c] > 0.f) p = __expf(S[row][c] - rmax) * inv;
    P[row][c] = f2bf(p);
  }
  __syncthreads();

  // phase 3: O = P * V  (wave -> 16 output dims)
  v8f o = z;
  const bf16* Pp = &P[0][0];
  for (int k2 = 0; k2 < NPADV; k2 += 32) {
    v16bf ap = load_a_frag(Pp, NPADV, 0, k2, lane);
    v16bf bv = load_b_frag(VT, NPADV, wave * 16, k2, lane);
    o = wmma_bf16(ap, bv, o);
  }
#pragma unroll
  for (int i = 0; i < 8; ++i) {
    int rr = hv * 8 + i;
    int q2 = qt * 16 + rr;
    if (q2 < NN)
      y_bf[((size_t)(b * NN + q2)) * CC + h * HD + wave * 16 + r] = f2bf(o[i]);
  }
}

// ---------------------------------------------------------------- pool+final LN
__global__ void __launch_bounds__(256) pool_final_kernel(
    const float* __restrict__ x3, const float* __restrict__ maskf,
    const float* __restrict__ og, const float* __restrict__ ob,
    float* __restrict__ out) {
  __shared__ float pooled[CC];
  __shared__ float red[256];
  const int b = blockIdx.x, tid = threadIdx.x;
  float ms = 0.f;
  for (int n = tid; n < NN; n += 256) ms += maskf[(size_t)b * NN + n];
  red[tid] = ms; __syncthreads();
  for (int st = 128; st > 0; st >>= 1) { if (tid < st) red[tid] += red[tid + st]; __syncthreads(); }
  float msum = red[0]; __syncthreads();
  for (int c = tid; c < CC; c += 256) {
    float acc = 0.f;
    for (int n = 0; n < NN; ++n)
      acc += maskf[(size_t)b * NN + n] * x3[((size_t)(b * NN + n)) * CC + c];
    pooled[c] = acc / msum;
  }
  __syncthreads();
  float s = 0.f, s2 = 0.f;
  for (int c = tid; c < CC; c += 256) { float v = pooled[c]; s += v; s2 += v * v; }
  red[tid] = s; __syncthreads();
  for (int st = 128; st > 0; st >>= 1) { if (tid < st) red[tid] += red[tid + st]; __syncthreads(); }
  float mean = red[0] / (float)CC; __syncthreads();
  red[tid] = s2; __syncthreads();
  for (int st = 128; st > 0; st >>= 1) { if (tid < st) red[tid] += red[tid + st]; __syncthreads(); }
  float var = red[0] / (float)CC - mean * mean;
  float inv = rsqrtf(var + 1e-6f);
  for (int c = tid; c < CC; c += 256)
    out[(size_t)b * CC + c] = (pooled[c] - mean) * inv * og[c] + ob[c];
}

// ---------------------------------------------------------------- launcher
extern "C" void kernel_launch(void* const* d_in, const int* in_sizes, int n_in,
                              void* d_out, int out_size, void* d_ws, size_t ws_size,
                              hipStream_t stream) {
  (void)in_sizes; (void)n_in; (void)out_size; (void)ws_size;
  const float* last_x  = (const float*)d_in[0];
  const float* ascores = (const float*)d_in[1];
  const float* n1g     = (const float*)d_in[2];
  const float* n1b     = (const float*)d_in[3];
  const float* qkv_w   = (const float*)d_in[4];
  const float* qkv_b   = (const float*)d_in[5];
  const float* proj_w  = (const float*)d_in[6];
  const float* proj_b  = (const float*)d_in[7];
  const float* n2g     = (const float*)d_in[8];
  const float* n2b     = (const float*)d_in[9];
  const float* fc1_w   = (const float*)d_in[10];
  const float* fc1_b   = (const float*)d_in[11];
  const float* fc2_w   = (const float*)d_in[12];
  const float* fc2_b   = (const float*)d_in[13];
  const float* bbng    = (const float*)d_in[14];
  const float* bbnb    = (const float*)d_in[15];
  const float* outg    = (const float*)d_in[16];
  const float* outb    = (const float*)d_in[17];
  float* out = (float*)d_out;

  char* ws = (char*)d_ws;
  size_t off = 0;
  auto alloc = [&](size_t bytes) -> char* {
    char* p = ws + off;
    off += (bytes + 255) & ~(size_t)255;
    return p;
  };
  float* maskf = (float*)alloc((size_t)MTOK * 4);
  bf16*  h_bf  = (bf16*)alloc((size_t)MTOK * CC * 2);
  char*  reg0  = alloc((size_t)MTOK * 3 * CC * 2);        // qkv_bf
  char*  reg0b = alloc((size_t)BB * HH * NPAD * HD * 2);  // Qb
  char*  reg1  = alloc((size_t)BB * HH * NPAD * HD * 2);  // Kb
  char*  reg1b = alloc((size_t)BB * HH * HD * NPADV * 2); // VTb
  bf16*  y_bf  = (bf16*)alloc((size_t)MTOK * CC * 2);
  float* x1    = (float*)alloc((size_t)MTOK * CC * 4);
  float* x2    = (float*)alloc((size_t)MTOK * CC * 4);
  bf16*  wq    = (bf16*)alloc((size_t)3 * CC * CC * 2);
  bf16*  wp    = (bf16*)alloc((size_t)CC * CC * 2);
  bf16*  w1    = (bf16*)alloc((size_t)4 * CC * CC * 2);
  bf16*  w2    = (bf16*)alloc((size_t)4 * CC * CC * 2);

  bf16* qkv_bf = (bf16*)reg0;
  bf16* Qb     = (bf16*)reg0b;
  bf16* Kb     = (bf16*)reg1;
  bf16* VTb    = (bf16*)reg1b;
  // deterministic buffer aliasing (lifetimes proven disjoint):
  bf16*  m_bf   = h_bf;          // h_bf dead after QKV GEMM
  bf16*  act_bf = (bf16*)reg0;   // qkv_bf+Qb dead after attention (114MB >= 113MB)
  float* x3     = (float*)reg1;  // Kb+VTb dead after attention (59MB >= 57MB)

  f32_to_bf16_kernel<<<(3 * CC * CC + 255) / 256, 256, 0, stream>>>(qkv_w, wq, 3 * CC * CC);
  f32_to_bf16_kernel<<<(CC * CC + 255) / 256, 256, 0, stream>>>(proj_w, wp, CC * CC);
  f32_to_bf16_kernel<<<(4 * CC * CC + 255) / 256, 256, 0, stream>>>(fc1_w, w1, 4 * CC * CC);
  f32_to_bf16_kernel<<<(4 * CC * CC + 255) / 256, 256, 0, stream>>>(fc2_w, w2, 4 * CC * CC);

  prune_kernel<<<BB, 256, 0, stream>>>(ascores, maskf);
  ln_kernel<<<MTOK, 256, 0, stream>>>(last_x, n1g, n1b, h_bf, nullptr);
  gemm_bf16_kernel<<<dim3((3 * CC) / 256, MTOK / 32), 128, 0, stream>>>(
      h_bf, wq, qkv_b, nullptr, nullptr, qkv_bf, 3 * CC, CC, 0);
  prep_attn_kernel<<<BB * HH, 256, 0, stream>>>(qkv_bf, Qb, Kb, VTb);
  attn_kernel<<<dim3(NKT, HH, BB), 128, 0, stream>>>(Qb, Kb, VTb, maskf, y_bf);
  gemm_bf16_kernel<<<dim3(CC / 256, MTOK / 32), 128, 0, stream>>>(
      y_bf, wp, proj_b, last_x, x1, nullptr, CC, CC, 0);
  ln_kernel<<<MTOK, 256, 0, stream>>>(x1, n2g, n2b, m_bf, nullptr);
  gemm_bf16_kernel<<<dim3((4 * CC) / 256, MTOK / 32), 128, 0, stream>>>(
      m_bf, w1, fc1_b, nullptr, nullptr, act_bf, 4 * CC, CC, 1);
  gemm_bf16_kernel<<<dim3(CC / 256, MTOK / 32), 128, 0, stream>>>(
      act_bf, w2, fc2_b, x1, x2, nullptr, CC, 4 * CC, 0);
  ln_kernel<<<MTOK, 256, 0, stream>>>(x2, bbng, bbnb, nullptr, x3);
  pool_final_kernel<<<BB, 256, 0, stream>>>(x3, maskf, outg, outb, out);
}
